// Codebook_20890720928571
// MI455X (gfx1250) — matrix operand order repaired and checked by reading, not compiled, not run on hardware
//
#include <hip/hip_runtime.h>

typedef __attribute__((ext_vector_type(16))) __bf16 v16bf;
typedef __attribute__((ext_vector_type(8)))  float  v8f;

typedef unsigned short u16;
typedef unsigned int   u32;

#define DDIM   256
#define MROWS  4096
#define NROWS  16384            // B*N = 16*1024
#define SPLITS 4
#define COLS_PER_BLOCK (MROWS / SPLITS)        // 1024
#define TILES_PER_BLOCK (COLS_PER_BLOCK / 16)  // 64

union Frag {
    v16bf v;
    uint4 q[2];
};

__device__ __forceinline__ u16 f2bf(float f) {
    u32 u = __float_as_uint(f);
    u += 0x7FFFu + ((u >> 16) & 1u);   // round-to-nearest-even
    return (u16)(u >> 16);
}

// Async global->LDS copy, 16 bytes per lane; tracked by ASYNCcnt (gfx1250).
__device__ __forceinline__ void async_ld_b128(void* lds_dst, const void* gsrc) {
    u32 ldsoff = (u32)(uintptr_t)lds_dst;   // low 32b of generic addr == LDS byte offset
    asm volatile("global_load_async_to_lds_b128 %0, %1, off"
                 :: "v"(ldsoff), "v"(gsrc)
                 : "memory");
}
__device__ __forceinline__ void wait_asynccnt0() {
    asm volatile("s_wait_asynccnt 0x0" ::: "memory");
}

// ---------------- Kernel 1: encode fp32 -> bf16 (elementwise, vectorized) ----
__global__ __launch_bounds__(256) void k_convE(const float* __restrict__ in,
                                               u16* __restrict__ out, int n4) {
    int i = blockIdx.x * 256 + threadIdx.x;
    if (i < n4) {
        float4 f = ((const float4*)in)[i];
        u32 lo = (u32)f2bf(f.x) | ((u32)f2bf(f.y) << 16);
        u32 hi = (u32)f2bf(f.z) | ((u32)f2bf(f.w) << 16);
        ((uint2*)out)[i] = make_uint2(lo, hi);
    }
}

// ---------------- Kernel 2: templat fp32 -> bf16 + |t|^2 per row -------------
__global__ __launch_bounds__(256) void k_convT(const float* __restrict__ T,
                                               u16* __restrict__ Tbf,
                                               float* __restrict__ tsq) {
    const int lane = threadIdx.x & 31;
    const int wave = threadIdx.x >> 5;
    const int row  = blockIdx.x * 8 + wave;          // 512 blocks * 8 rows
    const float* rp = T + row * DDIM + lane * 8;
    float4 f0 = *(const float4*)rp;
    float4 f1 = *(const float4*)(rp + 4);
    float ss = f0.x*f0.x + f0.y*f0.y + f0.z*f0.z + f0.w*f0.w
             + f1.x*f1.x + f1.y*f1.y + f1.z*f1.z + f1.w*f1.w;
    uint4 o;
    o.x = (u32)f2bf(f0.x) | ((u32)f2bf(f0.y) << 16);
    o.y = (u32)f2bf(f0.z) | ((u32)f2bf(f0.w) << 16);
    o.z = (u32)f2bf(f1.x) | ((u32)f2bf(f1.y) << 16);
    o.w = (u32)f2bf(f1.z) | ((u32)f2bf(f1.w) << 16);
    *(uint4*)(Tbf + row * DDIM + lane * 8) = o;
#pragma unroll
    for (int m = 16; m > 0; m >>= 1) ss += __shfl_xor(ss, m, 32);
    if (lane == 0) tsq[row] = ss;
}

// ---------------- Kernel 3: bf16 WMMA GEMM + fused partial argmax ------------
__global__ __launch_bounds__(256) void k_gemm_argmax(
        const u16* __restrict__ Ebf, const u16* __restrict__ Tbf,
        const float* __restrict__ tsq,
        float* __restrict__ part_score, int* __restrict__ part_idx) {
    const int tid   = threadIdx.x;
    const int lane  = tid & 31;
    const int wave  = tid >> 5;
    const int ahalf = lane >> 4;                 // 0/1: which K-half this lane holds
    const int rowBase  = blockIdx.x * 128 + wave * 16;   // 128 row-chunks
    const int colBase0 = blockIdx.y * COLS_PER_BLOCK;    // SPLITS column splits

    __shared__ u16 bstage[2][16 * DDIM];         // double-buffered 16-col B tile (2 x 8KB)

    // Per-thread staging slot: row (tid>>4) of the tile, 16-element chunk (tid&15).
    const int srow = tid >> 4;
    const int scol = (tid & 15) * 16;

    // ---- A fragments (16 rows x 256 K) resident in VGPRs for the whole kernel
    const u16* ap = Ebf + (rowBase + (lane & 15)) * DDIM;
    Frag afr[8];
#pragma unroll
    for (int kc = 0; kc < 8; ++kc) {
        afr[kc].q[0] = *(const uint4*)(ap + kc * 32 + ahalf * 8);
        afr[kc].q[1] = *(const uint4*)(ap + kc * 32 + 16 + ahalf * 8);
    }

    float best[8];
    int   bidx[8];
#pragma unroll
    for (int j = 0; j < 8; ++j) { best[j] = -3.4e38f; bidx[j] = 0; }

    // ---- stage B tile 0 via async global->LDS (ASYNCcnt path)
    {
        const u16* tp = Tbf + (colBase0 + srow) * DDIM + scol;
        u16* dp = &bstage[0][srow * DDIM + scol];
        async_ld_b128(dp,     tp);
        async_ld_b128(dp + 8, tp + 8);
    }
    wait_asynccnt0();
    __syncthreads();

    for (int t = 0; t < TILES_PER_BLOCK; ++t) {
        const int cur = t & 1, nxt = cur ^ 1;
        if (t + 1 < TILES_PER_BLOCK) {           // async prefetch of next tile
            const u16* tp = Tbf + (colBase0 + (t + 1) * 16 + srow) * DDIM + scol;
            u16* dp = &bstage[nxt][srow * DDIM + scol];
            async_ld_b128(dp,     tp);
            async_ld_b128(dp + 8, tp + 8);
        }

        v8f acc = {0.f, 0.f, 0.f, 0.f, 0.f, 0.f, 0.f, 0.f};
        const u16* bp = &bstage[cur][(lane & 15) * DDIM];
#pragma unroll
        for (int kc = 0; kc < 8; ++kc) {
            Frag bfr;
            bfr.q[0] = *(const uint4*)(bp + kc * 32 + ahalf * 8);
            bfr.q[1] = *(const uint4*)(bp + kc * 32 + 16 + ahalf * 8);
            acc = __builtin_amdgcn_wmma_f32_16x16x32_bf16(
                    false, afr[kc].v, false, bfr.v, (short)0, acc, false, false);
        }

        const int col = colBase0 + t * 16 + (lane & 15);
        const float tq = tsq[col];
#pragma unroll
        for (int j = 0; j < 8; ++j) {
            float s = 2.0f * acc[j] - tq;        // |e|^2 dropped (row-constant)
            if (s > best[j]) { best[j] = s; bidx[j] = col; }
        }

        wait_asynccnt0();                        // prefetch of t+1 has landed in LDS
        __syncthreads();
    }

    // ---- butterfly argmax across the 16 lanes that share each row set
#pragma unroll
    for (int m = 1; m < 16; m <<= 1) {
#pragma unroll
        for (int j = 0; j < 8; ++j) {
            float ob = __shfl_xor(best[j], m, 32);
            int   oi = __shfl_xor(bidx[j], m, 32);
            if (ob > best[j] || (ob == best[j] && oi < bidx[j])) {
                best[j] = ob; bidx[j] = oi;
            }
        }
    }
    if ((lane & 15) == 0) {                      // lane 0 -> rows 0..7, lane 16 -> rows 8..15
#pragma unroll
        for (int j = 0; j < 8; ++j) {
            const int row = rowBase + ahalf * 8 + j;
            part_score[row * SPLITS + blockIdx.y] = best[j];
            part_idx[row * SPLITS + blockIdx.y]   = bidx[j];
        }
    }
}

// ---------------- Kernel 4: combine splits, emit zidx, gather quant ----------
__global__ __launch_bounds__(256) void k_combine_gather(
        const float* __restrict__ part_score, const int* __restrict__ part_idx,
        const float* __restrict__ templat, float* __restrict__ out) {
    __shared__ int sidx[16];
    const int tid = threadIdx.x;
    const int rowBase = blockIdx.x * 16;
    if (tid < 16) {
        const int row = rowBase + tid;
        float b = -3.4e38f; int bi = 0;
        for (int s = 0; s < SPLITS; ++s) {       // splits are in increasing col order
            float v = part_score[row * SPLITS + s];
            int   i = part_idx[row * SPLITS + s];
            if (v > b || (v == b && i < bi)) { b = v; bi = i; }
        }
        sidx[tid] = bi;
        out[(size_t)NROWS * DDIM + row] = (float)bi;   // zidx region (fp32)
    }
    __syncthreads();
#pragma unroll
    for (int r = 0; r < 16; ++r)                 // coalesced fp32 gather from templat
        out[(size_t)(rowBase + r) * DDIM + tid] = templat[(size_t)sidx[r] * DDIM + tid];
}

// ---------------------------------------------------------------------------
extern "C" void kernel_launch(void* const* d_in, const int* in_sizes, int n_in,
                              void* d_out, int out_size, void* d_ws, size_t ws_size,
                              hipStream_t stream) {
    const float* encode  = (const float*)d_in[0];   // (16,1024,256) f32
    const float* templat = (const float*)d_in[1];   // (4096,256)    f32
    float* out = (float*)d_out;                     // quant (16*1024*256) ++ zidx (16384)

    char* ws = (char*)d_ws;
    u16*   Ebf    = (u16*)(ws + 0);                 //  8,388,608 B
    u16*   Tbf    = (u16*)(ws + 8388608);           //  2,097,152 B
    float* tsq    = (float*)(ws + 10485760);        //     16,384 B
    float* pscore = (float*)(ws + 10502144);        //    262,144 B
    int*   pidx   = (int*)(ws + 10764288);          //    262,144 B

    k_convE<<<dim3((NROWS * DDIM / 4 + 255) / 256), dim3(256), 0, stream>>>(
        encode, Ebf, NROWS * DDIM / 4);
    k_convT<<<dim3(MROWS / 8), dim3(256), 0, stream>>>(templat, Tbf, tsq);
    k_gemm_argmax<<<dim3(NROWS / 128, SPLITS), dim3(256), 0, stream>>>(
        Ebf, Tbf, tsq, pscore, pidx);
    k_combine_gather<<<dim3(NROWS / 16), dim3(256), 0, stream>>>(
        pscore, pidx, templat, out);
}
